// EnhancedGAT_29283087024639
// MI455X (gfx1250) — matrix operand order repaired and checked by reading, not compiled, not run on hardware
//
#include <hip/hip_runtime.h>
#include <hip/hip_bf16.h>
#include <math.h>

// ---------------------------------------------------------------------------
// EnhancedGAT for MI455X (gfx1250): bf16 WMMA GEMMs with async-to-LDS staging
// + f32 edge softmax/scatter (memory-roofline-bound part, kept in f32).
// ---------------------------------------------------------------------------

typedef __attribute__((ext_vector_type(16))) __bf16 v16bf;
typedef __attribute__((ext_vector_type(8)))  float  v8f;

#define GN 20000
#define GE 160000
#define NEG_SLOPE 0.2f

// ---------------- small helpers ----------------
static __device__ __forceinline__ unsigned f_ord(float f) {
  unsigned u = __float_as_uint(f);
  return (u & 0x80000000u) ? ~u : (u | 0x80000000u);   // order-preserving map
}
static __device__ __forceinline__ float ord_f(unsigned u) {
  u = (u & 0x80000000u) ? (u & 0x7fffffffu) : ~u;
  return __uint_as_float(u);
}
static __device__ __forceinline__ float elu1(float x) {
  return x > 0.f ? x : (expf(x) - 1.f);
}
// LDS byte address: ISA 10.2 — flat LDS-aperture address truncates to LDS_ADDR
static __device__ __forceinline__ unsigned lds_addr32(const void* p) {
  return (unsigned)(unsigned long long)p;
}

// ---------------- utility kernels ----------------
__global__ void k_zero_f32(float* p, long n) {
  long i = (long)blockIdx.x * blockDim.x + threadIdx.x;
  if (i < n) p[i] = 0.f;
}

__global__ void k_f32_to_bf16(const float* __restrict__ in, __bf16* __restrict__ out, long n) {
  long i = (long)blockIdx.x * blockDim.x + threadIdx.x;
  if (i < n) out[i] = (__bf16)in[i];
}

// ---------------- bf16 WMMA GEMM: C[M,Nt] = A[M,K] x B[K,Nt] (f32 out) ------
// Workgroup tile 128x64, K-step 32. 8 waves in a 4(M)x2(N) grid, each wave
// computes a 32x32 sub-tile with 2x2 v_wmma_f32_16x16x32_bf16.
// A tile is staged straight into LDS with GLOBAL_LOAD_ASYNC_TO_LDS_B128
// (ASYNCcnt), B tile is register-staged + transposed into LDS so both WMMA
// fragments become contiguous 16B ds_load_b128 reads.
#define TM 128
#define TN 64
#define TK 32
#define SA 40   // LDS row stride (elements) for A tile, conflict padding
#define SB 40   // LDS row stride (elements) for transposed B tile

__global__ __launch_bounds__(256) void k_gemm_bf16_wmma(
    const __bf16* __restrict__ A, const __bf16* __restrict__ B,
    float* __restrict__ C, int M, int Nt, int K, int ldc)
{
  __shared__ __bf16 As[TM * SA];        // [m][k]
  __shared__ __bf16 Bs[TN * SB];        // [n][k]  (transposed)

  const int tid  = threadIdx.x;
  const int lane = tid & 31;
  const int wid  = tid >> 5;
  const int half = lane >> 4;
  const int l16  = lane & 15;
  const int wm0  = (wid & 3) * 32;
  const int wn0  = (wid >> 2) * 32;
  const int m0   = blockIdx.x * TM;
  const int n0   = blockIdx.y * TN;

  // loader mapping: A -> 16 bf16/thread (async), B -> 8 bf16/thread (transpose)
  const int ar = tid >> 1;              // 0..127
  const int ah = (tid & 1) * 16;        // 0 or 16
  const int bk = tid >> 3;              // 0..31
  const int bc = (tid & 7) * 8;         // 0..56

  v8f acc[2][2] = {};

  const unsigned lds_a = lds_addr32(&As[ar * SA + ah]);
  const int gm = m0 + ar;

  for (int k0 = 0; k0 < K; k0 += TK) {
    // ---- stage A tile: async global -> LDS (32B per thread, 2 x B128) ----
    if (gm < M) {
      const unsigned long long ga =
          (unsigned long long)(const void*)(A + (long)gm * K + k0 + ah);
      // INST_OFFSET applies to both the global and the LDS address (ISA 10.x)
      asm volatile(
          "global_load_async_to_lds_b128 %0, %1, off\n\t"
          "global_load_async_to_lds_b128 %0, %1, off offset:16"
          :
          : "v"(lds_a), "v"(ga)
          : "memory");
    } else {
      // tail rows of the last M-tile: keep LDS benign
      const uint4 z = {0, 0, 0, 0};
      *(uint4*)&As[ar * SA + ah]     = z;
      *(uint4*)&As[ar * SA + ah + 8] = z;
    }

    // ---- stage B tile transposed into LDS ----
    {
      const uint4* pb = (const uint4*)(B + (long)(k0 + bk) * Nt + n0 + bc);
      uint4 bv = pb[0];
      if (k0 + TK < K)  // keep the B stream warm in GL2
        __builtin_prefetch(B + (long)(k0 + TK + bk) * Nt + n0 + bc, 0, 1);
      const __bf16* be = (const __bf16*)&bv;
      #pragma unroll
      for (int j = 0; j < 8; ++j) Bs[(bc + j) * SB + bk] = be[j];
    }

    // async A transfers must land before anyone reads the tile
    asm volatile("s_wait_asynccnt 0x0" ::: "memory");
    __syncthreads();

    // ---- load fragments per the CDNA5 16-bit WMMA VGPR layouts ----
    // A 16x32: lanes 0-15 row M=l16 hold K = {8*half+0..7} u {16+8*half+0..7}
    union Frag { uint4 u[2]; v16bf v; } af[2], bfg[2];
    #pragma unroll
    for (int i = 0; i < 2; ++i) {
      const __bf16* p = &As[(wm0 + i * 16 + l16) * SA + 8 * half];
      af[i].u[0] = *(const uint4*)p;
      af[i].u[1] = *(const uint4*)(p + 16);
    }
    // B 32x16: lane column N=l16 holds K = {8*half+0..7} u {16+8*half+0..7} in LDS-T
    #pragma unroll
    for (int j = 0; j < 2; ++j) {
      const __bf16* p = &Bs[(wn0 + j * 16 + l16) * SB + 8 * half];
      bfg[j].u[0] = *(const uint4*)p;
      bfg[j].u[1] = *(const uint4*)(p + 16);
    }

    #pragma unroll
    for (int i = 0; i < 2; ++i)
      #pragma unroll
      for (int j = 0; j < 2; ++j)
        acc[i][j] = __builtin_amdgcn_wmma_f32_16x16x32_bf16(
            false, af[i].v, false, bfg[j].v, (short)0, acc[i][j], false, false);
    __syncthreads();
  }

  // ---- store: C/D layout -> element r maps to M = r + 8*half, N = l16 ----
  #pragma unroll
  for (int i = 0; i < 2; ++i)
    #pragma unroll
    for (int j = 0; j < 2; ++j)
      #pragma unroll
      for (int r = 0; r < 8; ++r) {
        const int m = m0 + wm0 + i * 16 + r + 8 * half;
        const int n = n0 + wn0 + j * 16 + l16;
        if (m < M) C[(long)m * ldc + n] = acc[i][j][r];
      }
}

// ---------------- attention dot products: alpha_s/alpha_d [N,H] -------------
__global__ __launch_bounds__(256) void k_alpha_dot(
    const float* __restrict__ h, const float* __restrict__ a_src,
    const float* __restrict__ a_dst, float* __restrict__ as_out,
    float* __restrict__ ad_out, int Nn, int H, int C)
{
  const int lane = threadIdx.x & 31;
  const int wid  = threadIdx.x >> 5;
  const int g = blockIdx.x * 8 + wid;         // one wave per (node, head)
  if (g >= Nn * H) return;
  const int n = g / H, head = g % H;
  const float* hp = h + (long)n * H * C + (long)head * C;
  const float* sp = a_src + (long)head * C;
  const float* dp = a_dst + (long)head * C;
  float ss = 0.f, sd = 0.f;
  for (int c = lane; c < C; c += 32) {
    const float v = hp[c];
    ss += v * sp[c];
    sd += v * dp[c];
  }
  #pragma unroll
  for (int off = 16; off > 0; off >>= 1) {
    ss += __shfl_xor(ss, off, 32);
    sd += __shfl_xor(sd, off, 32);
  }
  if (lane == 0) { as_out[g] = ss; ad_out[g] = sd; }
}

// ---------------- edge pipeline ----------------
// edge id e in [0,E): real edge; e in [E, E+N): self loop (src=dst=e-E)
static __device__ __forceinline__ void edge_ends(const int* ei, long e, int E,
                                                 int& src, int& dst) {
  if (e < E) { src = ei[e]; dst = ei[E + e]; }
  else       { src = dst = (int)(e - E); }
}

__global__ __launch_bounds__(256) void k_edge_logits_max(
    const float* __restrict__ as, const float* __restrict__ ad,
    const int* __restrict__ ei, int E, long totEH, int H,
    float* __restrict__ e_buf, unsigned* __restrict__ emax)
{
  long t = (long)blockIdx.x * blockDim.x + threadIdx.x;
  if (t >= totEH) return;
  const int head = (int)(t % H);
  const long e = t / H;
  int src, dst; edge_ends(ei, e, E, src, dst);
  float v = as[(long)src * H + head] + ad[(long)dst * H + head];
  v = v > 0.f ? v : NEG_SLOPE * v;            // leaky_relu
  e_buf[t] = v;
  atomicMax(&emax[(long)dst * H + head], f_ord(v));
}

__global__ __launch_bounds__(256) void k_edge_exp_sum(
    const int* __restrict__ ei, int E, long totEH, int H,
    const unsigned* __restrict__ emax, float* __restrict__ e_buf,
    float* __restrict__ denom)
{
  long t = (long)blockIdx.x * blockDim.x + threadIdx.x;
  if (t >= totEH) return;
  const int head = (int)(t % H);
  const long e = t / H;
  int src, dst; edge_ends(ei, e, E, src, dst);
  const float m = ord_f(emax[(long)dst * H + head]);
  const float ee = expf(e_buf[t] - m);
  e_buf[t] = ee;
  atomicAdd(&denom[(long)dst * H + head], ee);
}

__global__ __launch_bounds__(256) void k_edge_norm(
    const int* __restrict__ ei, int E, long totEH, int H,
    const float* __restrict__ denom, float* __restrict__ e_buf)
{
  long t = (long)blockIdx.x * blockDim.x + threadIdx.x;
  if (t >= totEH) return;
  const int head = (int)(t % H);
  const long e = t / H;
  int src, dst; edge_ends(ei, e, E, src, dst);
  e_buf[t] = e_buf[t] / (denom[(long)dst * H + head] + 1e-16f);
}

__global__ __launch_bounds__(256) void k_edge_scatter(
    const float* __restrict__ h, const float* __restrict__ alpha,
    const int* __restrict__ ei, int E, long total, int H, int C,
    float* __restrict__ agg)
{
  long t = (long)blockIdx.x * blockDim.x + threadIdx.x;
  if (t >= total) return;
  const int c = (int)(t % C);
  const long eh = t / C;
  const int head = (int)(eh % H);
  const long e = eh / H;
  int src, dst; edge_ends(ei, e, E, src, dst);
  const float a = alpha[eh];
  atomicAdd(&agg[((long)dst * H + head) * C + c],
            h[((long)src * H + head) * C + c] * a);
}

// ---------------- epilogues ----------------
__global__ __launch_bounds__(256) void k_bias_elu_bf16(
    const float* __restrict__ in, const float* __restrict__ bias,
    __bf16* __restrict__ out, long rows, int cols, int ldo, int coff)
{
  long t = (long)blockIdx.x * blockDim.x + threadIdx.x;
  if (t >= rows * (long)cols) return;
  const long r = t / cols;
  const int c = (int)(t % cols);
  out[r * ldo + coff + c] = (__bf16)elu1(in[t] + bias[c]);
}

__global__ __launch_bounds__(256) void k_bias_add_f32(
    const float* __restrict__ in, const float* __restrict__ bias,
    float* __restrict__ out, long rows, int cols)
{
  long t = (long)blockIdx.x * blockDim.x + threadIdx.x;
  if (t >= rows * (long)cols) return;
  out[t] = in[t] + bias[(int)(t % cols)];
}

// ---------------------------------------------------------------------------
// host launcher
// ---------------------------------------------------------------------------
static inline unsigned cdivu(long a, long b) { return (unsigned)((a + b - 1) / b); }

extern "C" void kernel_launch(void* const* d_in, const int* in_sizes, int n_in,
                              void* d_out, int out_size, void* d_ws, size_t ws_size,
                              hipStream_t stream) {
  (void)in_sizes; (void)n_in; (void)out_size; (void)ws_size;
  const float* x      = (const float*)d_in[0];
  const int*   ei     = (const int*)  d_in[1];
  const float* skip_W = (const float*)d_in[2];
  const float* skip_b = (const float*)d_in[3];
  const float* W1     = (const float*)d_in[4];
  const float* as1    = (const float*)d_in[5];
  const float* ad1    = (const float*)d_in[6];
  const float* b1     = (const float*)d_in[7];
  const float* W2     = (const float*)d_in[8];
  const float* as2    = (const float*)d_in[9];
  const float* ad2    = (const float*)d_in[10];
  const float* b2     = (const float*)d_in[11];
  const float* W3     = (const float*)d_in[12];
  const float* as3    = (const float*)d_in[13];
  const float* ad3    = (const float*)d_in[14];
  const float* b3     = (const float*)d_in[15];
  float* out = (float*)d_out;

  const long N = GN, E = GE, Etot = GE + GN;

  // ---- carve workspace (all offsets 1 KiB aligned) ----
  char* w = (char*)d_ws;
  auto carve = [&](long bytes) -> char* {
    char* p = w;
    w += (bytes + 1023) & ~1023L;
    return p;
  };
  __bf16* x_bf    = (__bf16*)carve(N * 256 * 2);
  __bf16* wsk_bf  = (__bf16*)carve(256L * 1024 * 2);
  __bf16* w1_bf   = (__bf16*)carve(256L * 1024 * 2);
  __bf16* w2_bf   = (__bf16*)carve(1024L * 512 * 2);
  __bf16* w3_bf   = (__bf16*)carve(1536L * 128 * 2);
  __bf16* x1_bf   = (__bf16*)carve(N * 1024 * 2);
  __bf16* comb_bf = (__bf16*)carve(N * 1536 * 2);
  float*  bufA    = (float*)carve(N * 1024 * 4);   // h (GEMM output) per layer
  float*  bufB    = (float*)carve(N * 1024 * 4);   // aggregation per layer
  float*  al_s    = (float*)carve(N * 4 * 4);
  float*  al_d    = (float*)carve(N * 4 * 4);
  float*  emax    = (float*)carve(N * 4 * 4);      // used as unsigned (ordered)
  float*  denom   = (float*)carve(N * 4 * 4);
  float*  e_buf   = (float*)carve(Etot * 4 * 4);

  const dim3 blk(256);
  auto Z = [&](float* p, long n) {
    k_zero_f32<<<cdivu(n, 256), blk, 0, stream>>>(p, n);
  };

  // ---- precision converts ----
  k_f32_to_bf16<<<cdivu(N * 256, 256), blk, 0, stream>>>(x, x_bf, N * 256);
  k_f32_to_bf16<<<cdivu(256L * 1024, 256), blk, 0, stream>>>(skip_W, wsk_bf, 256L * 1024);
  k_f32_to_bf16<<<cdivu(256L * 1024, 256), blk, 0, stream>>>(W1, w1_bf, 256L * 1024);
  k_f32_to_bf16<<<cdivu(1024L * 512, 256), blk, 0, stream>>>(W2, w2_bf, 1024L * 512);
  k_f32_to_bf16<<<cdivu(1536L * 128, 256), blk, 0, stream>>>(W3, w3_bf, 1536L * 128);

  // ---- skip branch: identity = elu(x @ skip_W + b) -> combined[:, :1024] ----
  k_gemm_bf16_wmma<<<dim3(cdivu(N, TM), 1024 / TN), blk, 0, stream>>>(
      x_bf, wsk_bf, bufB, (int)N, 1024, 256, 1024);
  k_bias_elu_bf16<<<cdivu(N * 1024, 256), blk, 0, stream>>>(
      bufB, skip_b, comb_bf, N, 1024, 1536, 0);

  // helper: one GAT layer edge phase (h in bufA -> agg in bufB)
  auto gat_edges = [&](const float* aS, const float* aD, int H, int C) {
    const long EH = Etot * H;
    k_alpha_dot<<<cdivu(N * H, 8), blk, 0, stream>>>(bufA, aS, aD, al_s, al_d,
                                                     (int)N, H, C);
    Z(emax, N * H);
    Z(denom, N * H);
    Z(bufB, N * (long)H * C);
    k_edge_logits_max<<<cdivu(EH, 256), blk, 0, stream>>>(
        al_s, al_d, ei, (int)E, EH, H, e_buf, (unsigned*)emax);
    k_edge_exp_sum<<<cdivu(EH, 256), blk, 0, stream>>>(
        ei, (int)E, EH, H, (const unsigned*)emax, e_buf, denom);
    k_edge_norm<<<cdivu(EH, 256), blk, 0, stream>>>(
        ei, (int)E, EH, H, denom, e_buf);
    const long tot = EH * C;
    k_edge_scatter<<<cdivu(tot, 256), blk, 0, stream>>>(
        bufA, e_buf, ei, (int)E, tot, H, C, bufB);
  };

  // ---- GAT layer 1: 256 -> 4 x 256 ----
  k_gemm_bf16_wmma<<<dim3(cdivu(N, TM), 1024 / TN), blk, 0, stream>>>(
      x_bf, w1_bf, bufA, (int)N, 1024, 256, 1024);
  gat_edges(as1, ad1, 4, 256);
  k_bias_elu_bf16<<<cdivu(N * 1024, 256), blk, 0, stream>>>(
      bufB, b1, x1_bf, N, 1024, 1024, 0);                 // x1 (bf16)

  // ---- GAT layer 2: 1024 -> 2 x 256 ----
  k_gemm_bf16_wmma<<<dim3(cdivu(N, TM), 512 / TN), blk, 0, stream>>>(
      x1_bf, w2_bf, bufA, (int)N, 512, 1024, 512);
  gat_edges(as2, ad2, 2, 256);
  k_bias_elu_bf16<<<cdivu(N * 512, 256), blk, 0, stream>>>(
      bufB, b2, comb_bf, N, 512, 1536, 1024);             // combined[:, 1024:]

  // ---- GAT layer 3: 1536 -> 1 x 128 ----
  k_gemm_bf16_wmma<<<dim3(cdivu(N, TM), 128 / TN), blk, 0, stream>>>(
      comb_bf, w3_bf, bufA, (int)N, 128, 1536, 128);
  gat_edges(as3, ad3, 1, 128);
  k_bias_add_f32<<<cdivu(N * 128, 256), blk, 0, stream>>>(bufB, b3, out, N, 128);
}
